// JointAttention_22694607192350
// MI455X (gfx1250) — matrix operand order repaired and checked by reading, not compiled, not run on hardware
//
#include <hip/hip_runtime.h>
#include <math.h>

// ---------------- CDNA5 WMMA helpers (wave32, 16x16x32 f16 -> f32) ----------------
typedef __attribute__((ext_vector_type(16))) _Float16 v16h;
typedef __attribute__((ext_vector_type(8)))  _Float16 v8h;
typedef __attribute__((ext_vector_type(8)))  float    v8f;

#define WMMA(a, b, c) __builtin_amdgcn_wmma_f32_16x16x32_f16(false, (a), false, (b), (short)0, (c), false, false)

// A fragment (16x32, f16) from a row-major buffer buf[row][k], row stride `stride` halves.
__device__ __forceinline__ v16h frag_a(const _Float16* buf, int stride, int row0, int k0) {
  int lane = threadIdx.x & 31;
  int hlf = lane >> 4, r = lane & 15;
  const _Float16* p = buf + (size_t)(row0 + r) * stride + k0 + hlf * 8;
  v8h lo = *(const v8h*)p;
  v8h hi = *(const v8h*)(p + 16);
  v16h out;
#pragma unroll
  for (int i = 0; i < 8; ++i) { out[i] = lo[i]; out[i + 8] = hi[i]; }
  return out;
}

// B fragment (32x16, f16) from a transposed buffer bufT[n][k] (contiguous in k).
__device__ __forceinline__ v16h frag_b(const _Float16* bufT, int stride, int n0, int k0) {
  int lane = threadIdx.x & 31;
  int hlf = lane >> 4, r = lane & 15;
  const _Float16* p = bufT + (size_t)(n0 + r) * stride + k0 + hlf * 8;
  v8h lo = *(const v8h*)p;
  v8h hi = *(const v8h*)(p + 16);
  v16h out;
#pragma unroll
  for (int i = 0; i < 8; ++i) { out[i] = lo[i]; out[i + 8] = hi[i]; }
  return out;
}

__device__ __forceinline__ v16h elu1(v16h q) {  // elu(x)+1 elementwise
  v16h o;
#pragma unroll
  for (int i = 0; i < 16; ++i) {
    float v = (float)q[i];
    o[i] = (_Float16)(v > 0.f ? v + 1.f : __expf(v));
  }
  return o;
}

// Async 16-byte global->LDS copy (CDNA5, tracked by ASYNCcnt).
__device__ __forceinline__ void async_cp16(unsigned lds_off, const _Float16* gptr) {
  asm volatile("global_load_async_to_lds_b128 %0, %1, off"
               :: "v"(lds_off), "v"(gptr) : "memory");
}
__device__ __forceinline__ void wait_async0() {
  asm volatile("s_wait_asynccnt 0x0" ::: "memory");
}

// Problem constants
#define NB    2
#define NX    2048
#define NA    2048
#define DIMK  1024
#define HEADS 16
#define DH    64
#define NTOT  4096
#define SEGL  256
#define NSEG  16
#define QKVC  3072

// ---------------- fp32 -> f16 bulk convert (activations)
__global__ __launch_bounds__(256) void conv_f16(const float* __restrict__ src,
                                                _Float16* __restrict__ dst) {
  size_t i = ((size_t)blockIdx.x * 256 + threadIdx.x) * 8;
  float4 a = *(const float4*)(src + i);
  float4 b = *(const float4*)(src + i + 4);
  v8h o;
  o[0] = (_Float16)a.x; o[1] = (_Float16)a.y; o[2] = (_Float16)a.z; o[3] = (_Float16)a.w;
  o[4] = (_Float16)b.x; o[5] = (_Float16)b.y; o[6] = (_Float16)b.z; o[7] = (_Float16)b.w;
  *(v8h*)(dst + i) = o;
}

// ---------------- W[1024][3072] fp32 -> Wt[3072][1024] f16 (tiled transpose)
__global__ __launch_bounds__(256) void transpose_wt(const float* __restrict__ W,
                                                    _Float16* __restrict__ Wt) {
  __shared__ _Float16 tile[64][72];
  int kb = blockIdx.y * 64;
  int nb = blockIdx.x * 64;
  int t = threadIdx.x;
  int row = t >> 2;
#pragma unroll
  for (int c = 0; c < 16; c += 4) {
    int col = (t & 3) * 16 + c;
    float4 v = *(const float4*)(W + (size_t)(kb + row) * QKVC + nb + col);
    tile[col + 0][row] = (_Float16)v.x;
    tile[col + 1][row] = (_Float16)v.y;
    tile[col + 2][row] = (_Float16)v.z;
    tile[col + 3][row] = (_Float16)v.w;
  }
  __syncthreads();
#pragma unroll
  for (int c = 0; c < 2; ++c) {
    int col = (t & 3) * 16 + c * 8;
    *(v8h*)(Wt + (size_t)(nb + row) * DIMK + kb + col) = *(const v8h*)&tile[row][col];
  }
}

// ---------------- Phase 0: QKV GEMM: C[4096,3072] = Xh[4096,1024] @ Wt^T
//                  (f16 WMMA, double-buffered async-LDS pipeline)
__global__ __launch_bounds__(256) void qkv_gemm(const _Float16* __restrict__ Xh,
                                                const _Float16* __restrict__ Wt,
                                                float* __restrict__ C) {
  __shared__ _Float16 As[2][128][72];   // [stage][m][k]
  __shared__ _Float16 Bs[2][128][72];   // [stage][n][k]
  int m0 = blockIdx.y * 128;
  int n0 = blockIdx.x * 128;
  int t = threadIdx.x;
  int wid = t >> 5, lane = t & 31, hlf = lane >> 4, r = lane & 15;
  int rw = wid & 3, cw = wid >> 2;      // wave tile: 32 rows x 64 cols
  unsigned asb0 = (unsigned)(uintptr_t)&As[0][0][0];
  unsigned asb1 = (unsigned)(uintptr_t)&As[1][0][0];
  unsigned bsb0 = (unsigned)(uintptr_t)&Bs[0][0][0];
  unsigned bsb1 = (unsigned)(uintptr_t)&Bs[1][0][0];

  auto issue = [&](unsigned ab, unsigned bb, int kk) {
#pragma unroll
    for (int c = 0; c < 4; ++c) {
      int id = t + 256 * c;             // 0..1023 : 128 rows x 8 16B-chunks
      int row = id >> 3, c8 = id & 7;
      async_cp16(ab + row * 144 + c8 * 16, Xh + (size_t)(m0 + row) * DIMK + kk + c8 * 8);
      async_cp16(bb + row * 144 + c8 * 16, Wt + (size_t)(n0 + row) * DIMK + kk + c8 * 8);
    }
  };

  v8f acc[2][4] = {};
  issue(asb0, bsb0, 0);
  for (int kk = 0, buf = 0; kk < DIMK; kk += 64, buf ^= 1) {
    wait_async0();          // my copies for this chunk are the only outstanding ones
    __syncthreads();        // everyone's copies done + everyone done reading other stage
    if (kk + 64 < DIMK)
      issue(buf ? asb0 : asb1, buf ? bsb0 : bsb1, kk + 64);  // stream next while we compute
    const _Float16* Ab = &As[buf][0][0];
    const _Float16* Bb = &Bs[buf][0][0];
#pragma unroll
    for (int kc = 0; kc < 64; kc += 32) {
      v16h a0 = frag_a(Ab, 72, rw * 32, kc);
      v16h a1 = frag_a(Ab, 72, rw * 32 + 16, kc);
#pragma unroll
      for (int j = 0; j < 4; ++j) {
        v16h b = frag_b(Bb, 72, cw * 64 + j * 16, kc);
        acc[0][j] = WMMA(a0, b, acc[0][j]);
        acc[1][j] = WMMA(a1, b, acc[1][j]);
      }
    }
  }
#pragma unroll
  for (int ri = 0; ri < 2; ++ri)
#pragma unroll
    for (int j = 0; j < 4; ++j)
#pragma unroll
      for (int e = 0; e < 8; ++e) {
        int row = m0 + rw * 32 + ri * 16 + e + hlf * 8;
        int col = n0 + cw * 64 + j * 16 + r;
        C[(size_t)row * QKVC + col] = acc[ri][j][e];
      }
}

// ---------------- Phase 0.5: mh_rmsnorm + concat(a,x) + RoPE -> f16 Q,K,V [b][h][n][64]
__global__ __launch_bounds__(256) void norm_rope(const float* __restrict__ qkvx,
                                                 const float* __restrict__ qkva,
                                                 const float* __restrict__ gqx, const float* __restrict__ gkx,
                                                 const float* __restrict__ gqa, const float* __restrict__ gka,
                                                 _Float16* __restrict__ Q, _Float16* __restrict__ K,
                                                 _Float16* __restrict__ V) {
  int gw = (int)((blockIdx.x * 256u + threadIdx.x) >> 5);  // one wave per (b,h,n)
  int lane = threadIdx.x & 31;
  int n = gw & (NTOT - 1);
  int h = (gw >> 12) & (HEADS - 1);
  int b = gw >> 16;
  bool from_a = n < NA;
  int srow = b * 2048 + (from_a ? n : n - NA);
  const float* src = from_a ? qkva : qkvx;
  const float* gq = from_a ? gqa : gqx;
  const float* gk = from_a ? gka : gkx;
  int d0 = lane, d1 = lane + 32;
  size_t base = (size_t)srow * QKVC + h * DH;
  float q0 = src[base + d0],        q1 = src[base + d1];
  float k0 = src[base + 1024 + d0], k1 = src[base + 1024 + d1];
  float v0 = src[base + 2048 + d0], v1 = src[base + 2048 + d1];
  float nq = q0 * q0 + q1 * q1;
  float nk = k0 * k0 + k1 * k1;
#pragma unroll
  for (int off = 16; off > 0; off >>= 1) {
    nq += __shfl_xor(nq, off, 32);
    nk += __shfl_xor(nk, off, 32);
  }
  nq = fmaxf(sqrtf(nq), 1e-12f);
  nk = fmaxf(sqrtf(nk), 1e-12f);
  q0 = q0 / nq * gq[h * DH + d0] * 8.0f; q1 = q1 / nq * gq[h * DH + d1] * 8.0f;
  k0 = k0 / nk * gk[h * DH + d0] * 8.0f; k1 = k1 / nk * gk[h * DH + d1] * 8.0f;
  float freq = (float)n * __expf(-(float)lane * (9.210340371976184f / 32.0f));
  float sn, cs;
  __sincosf(freq, &sn, &cs);
  float qo0 = q0 * cs - q1 * sn, qo1 = q1 * cs + q0 * sn;
  float ko0 = k0 * cs - k1 * sn, ko1 = k1 * cs + k0 * sn;
  size_t ob = (((size_t)(b * HEADS + h) * NTOT) + n) * DH;
  Q[ob + d0] = (_Float16)qo0; Q[ob + d1] = (_Float16)qo1;
  K[ob + d0] = (_Float16)ko0; K[ob + d1] = (_Float16)ko1;
  V[ob + d0] = (_Float16)v0;  V[ob + d1] = (_Float16)v1;
}

// ---------------- Phase 1: per-segment memory contributions C_s = sk^T @ v (64x64), zc = sum sk
__global__ __launch_bounds__(256) void kv_contrib(const _Float16* __restrict__ K,
                                                  const _Float16* __restrict__ V,
                                                  float* __restrict__ Mbuf, float* __restrict__ zbuf) {
  __shared__ _Float16 skT[64][264];
  __shared__ _Float16 vT[64][264];
  int s = blockIdx.x, h = blockIdx.y, b = blockIdx.z;
  int t = threadIdx.x;
  size_t base = (((size_t)(b * HEADS + h) * NTOT) + s * SEGL) * DH;
  for (int i = t; i < SEGL * DH; i += 256) {
    int row = i >> 6, d = i & 63;
    float kv = (float)K[base + i];
    skT[d][row] = (_Float16)(kv > 0.f ? kv + 1.f : __expf(kv));
    vT[d][row] = V[base + i];
  }
  __syncthreads();
  size_t zb = (((size_t)(b * HEADS + h) * NSEG) + s) * DH;
  if (t < 64) {
    float acc = 0.f;
    for (int i = 0; i < SEGL; ++i) acc += (float)skT[t][i];
    zbuf[zb + t] = acc;
  }
  int wid = t >> 5, lane = t & 31, hlf = lane >> 4, r = lane & 15;
  size_t mb = (((size_t)(b * HEADS + h) * NSEG) + s) * (DH * DH);
#pragma unroll
  for (int tt = wid * 2; tt < wid * 2 + 2; ++tt) {
    int mbk = tt >> 2, nbk = tt & 3;
    v8f acc = {};
#pragma unroll
    for (int kc = 0; kc < SEGL; kc += 32) {
      v16h a = frag_a(&skT[0][0], 264, mbk * 16, kc);
      v16h bb = frag_b(&vT[0][0], 264, nbk * 16, kc);
      acc = WMMA(a, bb, acc);
    }
#pragma unroll
    for (int e = 0; e < 8; ++e) {
      int row = mbk * 16 + e + hlf * 8;  // d index
      int col = nbk * 16 + r;            // e index
      Mbuf[mb + (size_t)row * DH + col] = acc[e];
    }
  }
}

// ---------------- Phase 2: exclusive prefix scan over segments (M and z), per (b,h)
__global__ __launch_bounds__(256) void prefix_scan(float* __restrict__ Mbuf, float* __restrict__ zbuf) {
  int bh = blockIdx.x;
  int t = threadIdx.x;
  size_t mb = (size_t)bh * NSEG * (DH * DH);
  for (int e = t; e < DH * DH; e += 256) {
    float carry = 0.f;
#pragma unroll
    for (int s = 0; s < NSEG; ++s) {
      float v = Mbuf[mb + (size_t)s * (DH * DH) + e];
      Mbuf[mb + (size_t)s * (DH * DH) + e] = carry;
      carry += v;
    }
  }
  size_t zb = (size_t)bh * NSEG * DH;
  if (t < DH) {
    float carry = 0.f;
#pragma unroll
    for (int s = 0; s < NSEG; ++s) {
      float v = zbuf[zb + s * DH + t];
      zbuf[zb + s * DH + t] = carry;
      carry += v;
    }
  }
}

// ---------------- Phase 3: per-segment attention: softmax(QK^T)V, sq@M / (sq.z), gate blend
__global__ __launch_bounds__(256) void attn_out(const _Float16* __restrict__ Q,
                                                const _Float16* __restrict__ K,
                                                const _Float16* __restrict__ V,
                                                const float* __restrict__ Mbuf,
                                                const float* __restrict__ zbuf,
                                                const float* __restrict__ beta,
                                                float* __restrict__ out) {
  __shared__ _Float16 vT[64][264];       // v transposed: [e][seq]
  __shared__ _Float16 Mt[64][72];        // M transposed f16: [e][d]
  __shared__ float zsh[64];
  __shared__ _Float16 Pst[8][16][264];   // per-wave softmax strip
  int s = blockIdx.x, h = blockIdx.y, b = blockIdx.z;
  int t = threadIdx.x;
  size_t base = (((size_t)(b * HEADS + h) * NTOT) + s * SEGL) * DH;
  const _Float16* Qseg = Q + base;
  const _Float16* Kseg = K + base;
  __builtin_prefetch(Qseg + (size_t)(t) * DH, 0, 1);
  __builtin_prefetch(Kseg + (size_t)(t) * DH, 0, 1);
  for (int i = t; i < SEGL * DH; i += 256) {
    int row = i >> 6, d = i & 63;
    vT[d][row] = V[base + i];
  }
  size_t mb = (((size_t)(b * HEADS + h) * NSEG) + s) * (DH * DH);
  for (int i = t; i < DH * DH; i += 256) {
    int d = i >> 6, e = i & 63;
    Mt[e][d] = (_Float16)Mbuf[mb + i];
  }
  size_t zb = (((size_t)(b * HEADS + h) * NSEG) + s) * DH;
  if (t < 64) zsh[t] = zbuf[zb + t];
  __syncthreads();

  int wid = t >> 5, lane = t & 31, hlf = lane >> 4, r = lane & 15;
  float gate = 1.f / (1.f + __expf(-beta[h]));
  const float scale = 0.125f;  // DH^-0.5

  for (int rit = 0; rit < 2; ++rit) {
    int rbi = wid + rit * 8;      // row block (16 rows) of the 256-row segment
    int row0 = rbi * 16;
    int cbN = rbi | 1;            // last score col-tile needed (round K to 32)
    v16h aq0 = frag_a(&Qseg[0], DH, row0, 0);
    v16h aq1 = frag_a(&Qseg[0], DH, row0, 32);
    v8f acc[16];
    for (int cb = 0; cb <= cbN; ++cb) {
      v8f a0 = {};
      v16h b0 = frag_b(&Kseg[0], DH, cb * 16, 0);
      a0 = WMMA(aq0, b0, a0);
      v16h b1 = frag_b(&Kseg[0], DH, cb * 16, 32);
      a0 = WMMA(aq1, b1, a0);
      acc[cb] = a0;
    }
#pragma unroll
    for (int j = 0; j < 8; ++j) {
      int lrow = row0 + j + hlf * 8;   // local (in-segment) row for this element
      float mx = -3.0e38f;
      for (int cb = 0; cb <= cbN; ++cb) {
        int col = cb * 16 + r;
        float sc = acc[cb][j] * scale;
        if (col > lrow) sc = -3.0e38f;
        acc[cb][j] = sc;
        mx = fmaxf(mx, sc);
      }
#pragma unroll
      for (int off = 1; off < 16; off <<= 1) mx = fmaxf(mx, __shfl_xor(mx, off, 32));
      float sum = 0.f;
      for (int cb = 0; cb <= cbN; ++cb) {
        float e = __expf(acc[cb][j] - mx);
        acc[cb][j] = e;
        sum += e;
      }
#pragma unroll
      for (int off = 1; off < 16; off <<= 1) sum += __shfl_xor(sum, off, 32);
      float inv = 1.f / sum;
      int lr = j + hlf * 8;
      for (int cb = 0; cb <= cbN; ++cb)
        Pst[wid][lr][cb * 16 + r] = (_Float16)(acc[cb][j] * inv);
    }
    __syncthreads();
    v8f accL[4] = {}, accM[4] = {};
    int kcMax = (cbN + 1) * 16;
    for (int kc = 0; kc < kcMax; kc += 32) {
      v16h ap = frag_a(&Pst[wid][0][0], 264, 0, kc);
#pragma unroll
      for (int cb = 0; cb < 4; ++cb) {
        v16h bv = frag_b(&vT[0][0], 264, cb * 16, kc);
        accL[cb] = WMMA(ap, bv, accL[cb]);
      }
    }
#pragma unroll
    for (int kc = 0; kc < 64; kc += 32) {
      v16h asq = elu1(frag_a(&Qseg[0], DH, row0, kc));
#pragma unroll
      for (int cb = 0; cb < 4; ++cb) {
        v16h bm = frag_b(&Mt[0][0], 72, cb * 16, kc);
        accM[cb] = WMMA(asq, bm, accM[cb]);
      }
    }
    float den[8];
#pragma unroll
    for (int j = 0; j < 8; ++j) {
      int lrow = row0 + j + hlf * 8;
      float p = 0.f;
#pragma unroll
      for (int tt = 0; tt < 4; ++tt) {
        int d = r + tt * 16;
        float qv = (float)Qseg[(size_t)lrow * DH + d];
        float sq = qv > 0.f ? qv + 1.f : __expf(qv);
        p += sq * zsh[d];
      }
#pragma unroll
      for (int off = 1; off < 16; off <<= 1) p += __shfl_xor(p, off, 32);
      den[j] = p + 1e-6f;
    }
#pragma unroll
    for (int cb = 0; cb < 4; ++cb)
#pragma unroll
      for (int j = 0; j < 8; ++j) {
        int n = s * SEGL + row0 + j + hlf * 8;
        int d = cb * 16 + r;
        float val = gate * (accM[cb][j] / den[j]) + (1.f - gate) * accL[cb][j];
        size_t addr;
        if (n < NA)
          addr = (size_t)NB * NX * (HEADS * DH) +
                 ((size_t)(b * NA + n) * (HEADS * DH) + h * DH + d);
        else
          addr = (size_t)(b * NX + (n - NA)) * (HEADS * DH) + h * DH + d;
        out[addr] = val;
      }
    __syncthreads();
  }
}

// ---------------- Host launcher ----------------
extern "C" void kernel_launch(void* const* d_in, const int* in_sizes, int n_in,
                              void* d_out, int out_size, void* d_ws, size_t ws_size,
                              hipStream_t stream) {
  const float* x    = (const float*)d_in[0];
  const float* a    = (const float*)d_in[1];
  const float* wx   = (const float*)d_in[2];
  const float* wa   = (const float*)d_in[3];
  const float* gqx  = (const float*)d_in[4];
  const float* gkx  = (const float*)d_in[5];
  const float* gqa  = (const float*)d_in[6];
  const float* gka  = (const float*)d_in[7];
  const float* beta = (const float*)d_in[8];

  char* ws = (char*)d_ws;
  const size_t qkvBytes  = (size_t)4096 * QKVC * sizeof(float);   // 50,331,648
  const size_t qkvhBytes = (size_t)NB * HEADS * NTOT * DH * 2;    // 16,777,216
  const size_t mBytes    = (size_t)NB * HEADS * NSEG * DH * DH * sizeof(float);
  const size_t zBytes    = (size_t)NB * HEADS * NSEG * DH * sizeof(float);
  const size_t xhBytes   = (size_t)4096 * DIMK * 2;               // 8,388,608
  const size_t wtBytes   = (size_t)QKVC * DIMK * 2;               // 6,291,456

  size_t off = 0;
  float*    qkvx = (float*)(ws + off);    off += qkvBytes;
  float*    qkva = (float*)(ws + off);    off += qkvBytes;
  _Float16* Q    = (_Float16*)(ws + off); off += qkvhBytes;
  _Float16* K    = (_Float16*)(ws + off); off += qkvhBytes;
  _Float16* V    = (_Float16*)(ws + off); off += qkvhBytes;
  float*    Mbuf = (float*)(ws + off);    off += mBytes;
  float*    zbuf = (float*)(ws + off);    off += zBytes;
  _Float16* Xh   = (_Float16*)(ws + off); off += xhBytes;
  _Float16* Ah   = (_Float16*)(ws + off); off += xhBytes;
  _Float16* Wtx  = (_Float16*)(ws + off); off += wtBytes;
  _Float16* Wta  = (_Float16*)(ws + off); off += wtBytes;
  float* out = (float*)d_out;

  // f16 pre-conversion + weight transpose
  conv_f16<<<(4096 * DIMK) / 2048, 256, 0, stream>>>(x, Xh);
  conv_f16<<<(4096 * DIMK) / 2048, 256, 0, stream>>>(a, Ah);
  dim3 gt(QKVC / 64, DIMK / 64);
  transpose_wt<<<gt, 256, 0, stream>>>(wx, Wtx);
  transpose_wt<<<gt, 256, 0, stream>>>(wa, Wta);

  // QKV projections (WMMA, double-buffered async-LDS pipeline)
  dim3 gg(QKVC / 128, 4096 / 128);
  qkv_gemm<<<gg, 256, 0, stream>>>(Xh, Wtx, qkvx);
  qkv_gemm<<<gg, 256, 0, stream>>>(Ah, Wta, qkva);

  norm_rope<<<(NB * HEADS * NTOT * 32) / 256, 256, 0, stream>>>(
      qkvx, qkva, gqx, gkx, gqa, gka, Q, K, V);

  dim3 gs(NSEG, HEADS, NB);
  kv_contrib<<<gs, 256, 0, stream>>>(K, V, Mbuf, zbuf);
  prefix_scan<<<NB * HEADS, 256, 0, stream>>>(Mbuf, zbuf);
  attn_out<<<gs, 256, 0, stream>>>(Q, K, V, Mbuf, zbuf, beta, out);
}
